// TabRM_85229331022178
// MI455X (gfx1250) — compile-verified, matrix operand order checked by reading, not run on hardware
//
#include <hip/hip_runtime.h>

typedef _Float16 half_t;
typedef _Float16 v8h  __attribute__((ext_vector_type(8)));
typedef _Float16 v16h __attribute__((ext_vector_type(16)));
typedef float    v8f  __attribute__((ext_vector_type(8)));

#define BN_EPS 1e-5f

// ----------------------------------------------------------------------------
// CDNA5 async global->LDS copy (16B per lane), with portable fallback.
// Builtin signature (from compiler diagnostic): takes v4i* pointers.
// ----------------------------------------------------------------------------
__device__ __forceinline__ void async_copy_b128(const half_t* g, half_t* l)
{
#if defined(__gfx1250__) && __has_builtin(__builtin_amdgcn_global_load_async_to_lds_b128)
    typedef int v4i __attribute__((vector_size(16)));
    __builtin_amdgcn_global_load_async_to_lds_b128((v4i*)g, (v4i*)l, 0, 0);
#else
    *(v8h*)l = *(const v8h*)g;
#endif
}

__device__ __forceinline__ void async_wait()
{
#if defined(__gfx1250__) && __has_builtin(__builtin_amdgcn_s_wait_asynccnt)
    __builtin_amdgcn_s_wait_asynccnt(0);
#elif defined(__gfx1250__)
    asm volatile("s_wait_asynccnt 0x0" ::: "memory");
#endif
}

// ----------------------------------------------------------------------------
// f32 -> f16 conversion (inputs and weights, done once per launch)
// ----------------------------------------------------------------------------
__global__ void f32_to_f16(const float* __restrict__ src, half_t* __restrict__ dst,
                           size_t n)
{
    for (size_t i = (size_t)blockIdx.x * blockDim.x + threadIdx.x; i < n;
         i += (size_t)gridDim.x * blockDim.x)
        dst[i] = (half_t)src[i];
}

// ----------------------------------------------------------------------------
// WMMA GEMM: C[m][n] = act( sum_k A[m][k] * Bw[n][k] + bias[n] ), all f16 in,
// f32 accumulate, f16 out. 256 threads = 8 waves; workgroup tile 64x64; each
// wave computes 16(M) x 32(N) with two WMMAs sharing one A fragment.
// M,N multiples of 64; K multiple of 32 (true at all call sites).
// ----------------------------------------------------------------------------
template <bool RELU>
__global__ __launch_bounds__(256)
void wmma_gemm_bias(const half_t* __restrict__ A, const half_t* __restrict__ Bw,
                    const float* __restrict__ bias, half_t* __restrict__ C,
                    int M, int N, int K)
{
    __shared__ half_t As[64][32];
    __shared__ half_t Bs[64][32];
    const int tid  = threadIdx.x;
    const int lane = tid & 31;
    const int wave = tid >> 5;
    const int wm   = (wave >> 1) * 16;   // 0,16,32,48
    const int wn   = (wave & 1) * 32;    // 0,32
    const int bm   = blockIdx.y * 64;
    const int bn   = blockIdx.x * 64;
    const int g    = lane >> 4;
    const int l15  = lane & 15;
    // staging slice for this thread: one 8-half segment of each tile
    const int srow = tid >> 2;
    const int scol = (tid & 3) * 8;

    v8f acc0 = {}, acc1 = {};
    for (int k0 = 0; k0 < K; k0 += 32) {
        __syncthreads();
        async_copy_b128(&A [(size_t)(bm + srow) * K + k0 + scol], &As[srow][scol]);
        async_copy_b128(&Bw[(size_t)(bn + srow) * K + k0 + scol], &Bs[srow][scol]);
        async_wait();
        __syncthreads();
        // fragments per 16-bit 16x32 layout:
        // lane<16: row=lane, halves = K 0..7 then 16..23 ; lane>=16: K 8..15, 24..31
        v16h a, b0, b1;
        const v8h alo  = *(const v8h*)&As[wm + l15][g * 8];
        const v8h ahi  = *(const v8h*)&As[wm + l15][16 + g * 8];
        const v8h b0lo = *(const v8h*)&Bs[wn + l15][g * 8];
        const v8h b0hi = *(const v8h*)&Bs[wn + l15][16 + g * 8];
        const v8h b1lo = *(const v8h*)&Bs[wn + 16 + l15][g * 8];
        const v8h b1hi = *(const v8h*)&Bs[wn + 16 + l15][16 + g * 8];
        for (int i = 0; i < 8; ++i) {
            a[i]  = alo[i];  a[i + 8]  = ahi[i];
            b0[i] = b0lo[i]; b0[i + 8] = b0hi[i];
            b1[i] = b1lo[i]; b1[i + 8] = b1hi[i];
        }
        acc0 = __builtin_amdgcn_wmma_f32_16x16x32_f16(
            false, a, false, b0, (short)0, acc0, false, false);
        acc1 = __builtin_amdgcn_wmma_f32_16x16x32_f16(
            false, a, false, b1, (short)0, acc1, false, false);
    }
    // C/D layout: N = lane&15, M = r + 8*(lane>=16)
    const int n0 = bn + wn + l15;
    const int n1 = n0 + 16;
    const float bv0 = bias[n0];
    const float bv1 = bias[n1];
    for (int r = 0; r < 8; ++r) {
        const size_t row = (size_t)(bm + wm + r + 8 * g);
        float v0 = acc0[r] + bv0;
        float v1 = acc1[r] + bv1;
        if (RELU) { v0 = v0 > 0.f ? v0 : 0.f; v1 = v1 > 0.f ? v1 : 0.f; }
        C[row * N + n0] = (half_t)v0;
        C[row * N + n1] = (half_t)v1;
    }
}

// ----------------------------------------------------------------------------
// BatchNorm stats: one block per column; biased variance over M rows.
// ----------------------------------------------------------------------------
__global__ __launch_bounds__(256)
void bn_stats(const half_t* __restrict__ H, int M, int N,
              float* __restrict__ mean, float* __restrict__ var)
{
    __shared__ float s1[256], s2[256];
    const int col = blockIdx.x;
    float a1 = 0.f, a2 = 0.f;
    for (int r = threadIdx.x; r < M; r += 256) {
        float v = (float)H[(size_t)r * N + col];
        a1 += v; a2 += v * v;
    }
    s1[threadIdx.x] = a1; s2[threadIdx.x] = a2;
    __syncthreads();
    for (int s = 128; s > 0; s >>= 1) {
        if (threadIdx.x < s) {
            s1[threadIdx.x] += s1[threadIdx.x + s];
            s2[threadIdx.x] += s2[threadIdx.x + s];
        }
        __syncthreads();
    }
    if (threadIdx.x == 0) {
        float m = s1[0] / (float)M;
        mean[col] = m;
        var[col]  = s2[0] / (float)M - m * m;
    }
}

__global__ void bn_apply(const half_t* __restrict__ H, half_t* __restrict__ Y,
                         const float* __restrict__ mean, const float* __restrict__ var,
                         const float* __restrict__ g, const float* __restrict__ be,
                         int M, int N)
{
    const size_t total = (size_t)M * N;
    for (size_t i = (size_t)blockIdx.x * blockDim.x + threadIdx.x; i < total;
         i += (size_t)gridDim.x * blockDim.x) {
        int c = (int)(i % (size_t)N);
        float v = ((float)H[i] - mean[c]) * rsqrtf(var[c] + BN_EPS) * g[c] + be[c];
        Y[i] = (half_t)v;
    }
}

__global__ void row_norm2(const half_t* __restrict__ H, float* __restrict__ out,
                          int M, int N)
{
    for (int r = blockIdx.x * blockDim.x + threadIdx.x; r < M;
         r += gridDim.x * blockDim.x) {
        float s = 0.f;
        for (int c = 0; c < N; ++c) {
            float v = (float)H[(size_t)r * N + c];
            s += v * v;
        }
        out[r] = s;
    }
}

// ----------------------------------------------------------------------------
// Fused L2 distance (WMMA inner products) + streaming top-K=32.
// One workgroup (256 thr / 8 waves) per 16 queries; candidate tiles of 128.
// Candidate embeddings (67 MB) stay L2-resident across the 64 workgroups.
// ----------------------------------------------------------------------------
#define NCAND 65536
#define DK    512
__global__ __launch_bounds__(256)
void knn_topk(const half_t* __restrict__ Q, const half_t* __restrict__ Cemb,
              const float* __restrict__ qn, const float* __restrict__ cn,
              int* __restrict__ idx_out)
{
    __shared__ half_t Qs[16][DK];      // query tile, resident whole kernel
    __shared__ half_t Cs[128][32];     // candidate k-slab
    __shared__ float  Sc[16][128];     // score tile
    __shared__ float  kd[16][32];      // top-K distances
    __shared__ int    ki[16][32];      // top-K indices
    const int tid  = threadIdx.x;
    const int lane = tid & 31;
    const int wave = tid >> 5;
    const int g    = lane >> 4;
    const int l15  = lane & 15;
    const int qbase = blockIdx.x * 16;

    // stage query tile: 8192 halves = 1024 x 16B segments, 4 per thread
    for (int i = 0; i < 4; ++i) {
        int seg = i * 256 + tid;
        int m = seg >> 6, c = (seg & 63) * 8;
        async_copy_b128(&Q[(size_t)(qbase + m) * DK + c], &Qs[m][c]);
    }
    async_wait();
    if (tid < 16)
        for (int j = 0; j < 32; ++j) { kd[tid][j] = 3.0e38f; ki[tid][j] = 0; }
    float maxv = 3.0e38f; int maxp = 0;      // used by threads < 16 only

    for (int tile = 0; tile < NCAND / 128; ++tile) {
        const int cb = tile * 128;
        v8f acc = {};
        for (int ks = 0; ks < DK / 32; ++ks) {
            __syncthreads();
            const int k0 = ks * 32;
            // stage 128x32 candidate slab: 512 x 16B segments, 2 per thread
            for (int i = 0; i < 2; ++i) {
                int seg = i * 256 + tid;
                int r = seg >> 2, c = (seg & 3) * 8;
                async_copy_b128(&Cemb[(size_t)(cb + r) * DK + k0 + c], &Cs[r][c]);
            }
            async_wait();
            __syncthreads();
            v16h a, b;
            const v8h alo = *(const v8h*)&Qs[l15][k0 + g * 8];
            const v8h ahi = *(const v8h*)&Qs[l15][k0 + 16 + g * 8];
            const v8h blo = *(const v8h*)&Cs[wave * 16 + l15][g * 8];
            const v8h bhi = *(const v8h*)&Cs[wave * 16 + l15][16 + g * 8];
            for (int i = 0; i < 8; ++i) {
                a[i] = alo[i]; a[i + 8] = ahi[i];
                b[i] = blo[i]; b[i + 8] = bhi[i];
            }
            acc = __builtin_amdgcn_wmma_f32_16x16x32_f16(
                false, a, false, b, (short)0, acc, false, false);
        }
        // d2 = |q|^2 + |c|^2 - 2 q.c  into score tile
        const int col = wave * 16 + l15;
        const float cnv = cn[cb + col];
        for (int r = 0; r < 8; ++r) {
            int m = r + 8 * g;
            Sc[m][col] = qn[qbase + m] + cnv - 2.0f * acc[r];
        }
        __syncthreads();
        if (tid < 16) {
            for (int j = 0; j < 128; ++j) {
                float d = Sc[tid][j];
                if (d < maxv) {
                    kd[tid][maxp] = d; ki[tid][maxp] = cb + j;
                    maxv = kd[tid][0]; maxp = 0;
                    for (int t = 1; t < 32; ++t)
                        if (kd[tid][t] > maxv) { maxv = kd[tid][t]; maxp = t; }
                }
            }
        }
    }
    __syncthreads();
    if (tid < 16) {
        // ascending selection sort (matches top_k ordering), emit indices
        for (int j = 0; j < 32; ++j) {
            int best = j;
            for (int t = j + 1; t < 32; ++t)
                if (kd[tid][t] < kd[tid][best]) best = t;
            float td = kd[tid][j]; kd[tid][j] = kd[tid][best]; kd[tid][best] = td;
            int   ti = ki[tid][j]; ki[tid][j] = ki[tid][best]; ki[tid][best] = ti;
            idx_out[(qbase + tid) * 32 + j] = ki[tid][j];
        }
    }
}

// ----------------------------------------------------------------------------
// Build MLP input rows: concat(q[b], c_emb[idx[b][k]]) -> (32768 x 1024) f16
// ----------------------------------------------------------------------------
__global__ void gather_concat(const half_t* __restrict__ Q,
                              const half_t* __restrict__ Cemb,
                              const int* __restrict__ idx,
                              half_t* __restrict__ out)
{
    const size_t total = (size_t)32768 * 1024;
    for (size_t i = (size_t)blockIdx.x * blockDim.x + threadIdx.x; i < total;
         i += (size_t)gridDim.x * blockDim.x) {
        int r = (int)(i >> 10);
        int c = (int)(i & 1023);
        int b = r >> 5;
        half_t v;
        if (c < 512) v = Q[(size_t)b * 512 + c];
        else         v = Cemb[(size_t)idx[r] * 512 + (c - 512)];
        out[i] = v;
    }
}

// ----------------------------------------------------------------------------
// Output head: out[r] = dot(H[r,:1024], Wo) + bo   (wave-per-row reduction)
// ----------------------------------------------------------------------------
__global__ __launch_bounds__(256)
void head_dot(const half_t* __restrict__ H, const float* __restrict__ Wo,
              const float* __restrict__ bo, float* __restrict__ out)
{
    const int r    = blockIdx.x * 8 + (threadIdx.x >> 5);
    const int lane = threadIdx.x & 31;
    float s = 0.f;
    for (int j = lane; j < 1024; j += 32)
        s += (float)H[(size_t)r * 1024 + j] * Wo[j];
    for (int o = 16; o > 0; o >>= 1) s += __shfl_down(s, o, 32);
    if (lane == 0) out[r] = s + bo[0];
}

// ----------------------------------------------------------------------------
extern "C" void kernel_launch(void* const* d_in, const int* in_sizes, int n_in,
                              void* d_out, int out_size, void* d_ws, size_t ws_size,
                              hipStream_t stream)
{
    (void)in_sizes; (void)n_in; (void)out_size; (void)ws_size;
    const float* x   = (const float*)d_in[0];
    const float* cx  = (const float*)d_in[1];
    const float* W0  = (const float*)d_in[2];
    const float* b0  = (const float*)d_in[3];
    const float* g1  = (const float*)d_in[4];
    const float* be1 = (const float*)d_in[5];
    const float* W1  = (const float*)d_in[6];
    const float* b1  = (const float*)d_in[7];
    const float* W2  = (const float*)d_in[8];
    const float* b2  = (const float*)d_in[9];
    const float* g2  = (const float*)d_in[10];
    const float* be2 = (const float*)d_in[11];
    const float* Wm0 = (const float*)d_in[12];
    const float* bm0 = (const float*)d_in[13];
    const float* Wm1 = (const float*)d_in[14];
    const float* bm1 = (const float*)d_in[15];
    const float* Wo  = (const float*)d_in[16];
    const float* bo  = (const float*)d_in[17];
    float* out = (float*)d_out;

    char* ws = (char*)d_ws;
    size_t off = 0;
    auto alloc = [&](size_t bytes) -> char* {
        char* p = ws + off;
        off = (off + bytes + 255) & ~(size_t)255;
        return p;
    };
    half_t* qb0  = (half_t*)alloc(1024ull * 512 * 2);
    half_t* qb1  = (half_t*)alloc(1024ull * 512 * 2);
    half_t* cb0  = (half_t*)alloc(65536ull * 512 * 2);   // 64 MiB
    half_t* cb1  = (half_t*)alloc(65536ull * 512 * 2);   // 64 MiB
    float*  qn   = (float*)alloc(1024ull * 4);
    float*  cnr  = (float*)alloc(65536ull * 4);
    float*  mean = (float*)alloc(512ull * 4);
    float*  var  = (float*)alloc(512ull * 4);
    int*    idx  = (int*)alloc(1024ull * 32 * 4);
    half_t* x16   = (half_t*)alloc(1024ull * 96 * 2);
    half_t* cx16  = (half_t*)alloc(65536ull * 96 * 2);
    half_t* W0h   = (half_t*)alloc(512ull * 96 * 2);
    half_t* W1h   = (half_t*)alloc(512ull * 512 * 2);
    half_t* W2h   = (half_t*)alloc(512ull * 512 * 2);
    half_t* Wm0h  = (half_t*)alloc(1024ull * 1024 * 2);
    half_t* Wm1h  = (half_t*)alloc(1024ull * 1024 * 2);

    // ---- one-time f32 -> f16 conversions (inputs + weights)
    f32_to_f16<<<96,   256, 0, stream>>>(x,   x16,  1024ull * 96);
    f32_to_f16<<<6144, 256, 0, stream>>>(cx,  cx16, 65536ull * 96);
    f32_to_f16<<<48,   256, 0, stream>>>(W0,  W0h,  512ull * 96);
    f32_to_f16<<<256,  256, 0, stream>>>(W1,  W1h,  512ull * 512);
    f32_to_f16<<<256,  256, 0, stream>>>(W2,  W2h,  512ull * 512);
    f32_to_f16<<<1024, 256, 0, stream>>>(Wm0, Wm0h, 1024ull * 1024);
    f32_to_f16<<<1024, 256, 0, stream>>>(Wm1, Wm1h, 1024ull * 1024);

    // ---- embedder layer 0: Z @ W0^T + b0
    wmma_gemm_bias<false><<<dim3(8, 16),   256, 0, stream>>>(x16,  W0h, b0, qb0, 1024,  512, 96);
    wmma_gemm_bias<false><<<dim3(8, 1024), 256, 0, stream>>>(cx16, W0h, b0, cb0, 65536, 512, 96);
    // ---- BN1 (separate stats for q and c, as in reference)
    bn_stats<<<512, 256, 0, stream>>>(qb0, 1024, 512, mean, var);
    bn_apply<<<1024, 256, 0, stream>>>(qb0, qb1, mean, var, g1, be1, 1024, 512);
    bn_stats<<<512, 256, 0, stream>>>(cb0, 65536, 512, mean, var);
    bn_apply<<<8192, 256, 0, stream>>>(cb0, cb1, mean, var, g1, be1, 65536, 512);
    // ---- layer 1 (ReLU) and layer 2
    wmma_gemm_bias<true ><<<dim3(8, 16),   256, 0, stream>>>(qb1, W1h, b1, qb0, 1024,  512, 512);
    wmma_gemm_bias<true ><<<dim3(8, 1024), 256, 0, stream>>>(cb1, W1h, b1, cb0, 65536, 512, 512);
    wmma_gemm_bias<false><<<dim3(8, 16),   256, 0, stream>>>(qb0, W2h, b2, qb1, 1024,  512, 512);
    wmma_gemm_bias<false><<<dim3(8, 1024), 256, 0, stream>>>(cb0, W2h, b2, cb1, 65536, 512, 512);
    // ---- BN2 -> final embeddings (q_emb=qb0, c_emb=cb0)
    bn_stats<<<512, 256, 0, stream>>>(qb1, 1024, 512, mean, var);
    bn_apply<<<1024, 256, 0, stream>>>(qb1, qb0, mean, var, g2, be2, 1024, 512);
    bn_stats<<<512, 256, 0, stream>>>(cb1, 65536, 512, mean, var);
    bn_apply<<<8192, 256, 0, stream>>>(cb1, cb0, mean, var, g2, be2, 65536, 512);
    // ---- squared norms
    row_norm2<<<32,  256, 0, stream>>>(qb0, qn,  1024,  512);
    row_norm2<<<256, 256, 0, stream>>>(cb0, cnr, 65536, 512);
    // ---- fused WMMA distance + top-32
    knn_topk<<<64, 256, 0, stream>>>(qb0, cb0, qn, cnr, idx);
    // ---- gather concat(q, ctx) -> mlp input (reuse cb1)
    gather_concat<<<8192, 256, 0, stream>>>(qb0, cb0, idx, cb1);
    // ---- context MLP (reuse buffers: cb1 -> cb0 -> cb1)
    wmma_gemm_bias<true><<<dim3(16, 512), 256, 0, stream>>>(cb1, Wm0h, bm0, cb0, 32768, 1024, 1024);
    wmma_gemm_bias<true><<<dim3(16, 512), 256, 0, stream>>>(cb0, Wm1h, bm1, cb1, 32768, 1024, 1024);
    // ---- head
    head_dot<<<4096, 256, 0, stream>>>(cb1, Wo, bo, out);
}
